// DeeperGCN_LineGraph_16028817948902
// MI455X (gfx1250) — matrix operand order repaired and checked by reading, not compiled, not run on hardware
//
#include <hip/hip_runtime.h>

// ---------------- problem constants (match reference) ----------------
#define N_G      32768
#define E_G      131072
#define E_LG     262144
#define HDIM     128
#define NODE_DIM 32
#define EDGE_DIM 16
#define NBASIS   4
#define NLAYER   4
#define NBATCH   256
#define NTASKS   10

#define EPS_MSG  1e-7f
#define BN_INV   0.99999500003749968f   // 1/sqrt(1+1e-5)
#define NEG_BIG  -3.0e38f

typedef __attribute__((ext_vector_type(16))) _Float16 v16h;
typedef __attribute__((ext_vector_type(8)))  _Float16 v8h;
typedef __attribute__((ext_vector_type(8)))  float    v8f;

// =====================================================================
// WMMA fragment helpers (CDNA5 v_wmma_f32_16x16x32_f16, wave32; §7.12.2)
//  A 16x32 f16: lane(0-15)=row M, lane-half selects K sub-block; per lane
//    the 16 elements are two contiguous 16B runs: k0+half*8+[0..7] and
//    k0+16+half*8+[0..7]  -> two ds_load_b128.
//  B 32x16 f16: lane -> N=lane%16, K=(lane/16)*8+e%8+16*(e/8); pre-packed in
//    memory as 16 contiguous f16 per lane -> one 32B global load.
//  C/D 16x16 f32: VGPR r: N=lane%16, M=(lane/16)*8+r
// =====================================================================
__device__ inline v16h frag_a_lds(const _Float16* A, int ld, int k0, int lane) {
  const int half = lane >> 4, m = lane & 15;
  const v8h lo = *(const v8h*)(A + m * ld + k0 + half * 8);
  const v8h hi = *(const v8h*)(A + m * ld + k0 + 16 + half * 8);
  v16h a;
#pragma unroll
  for (int i = 0; i < 8; ++i) { a[i] = lo[i]; a[8 + i] = hi[i]; }
  return a;
}

__device__ inline v16h frag_b_packed(const _Float16* Wp, int frag, int lane) {
  return *(const v16h*)(Wp + (((size_t)frag * 32 + lane) << 4));
}

__device__ inline void atomicMaxF32(float* addr, float val) {
  int* ia = reinterpret_cast<int*>(addr);
  int cur = *ia;
  while (__int_as_float(cur) < val) {
    int prev = atomicCAS(ia, cur, __float_as_int(val));
    if (prev == cur) break;
    cur = prev;
  }
}

// =====================================================================
// Kernel 0: one-time weight pre-pack into B-fragment layout (f16).
// W row-major [Ktot,N] f32 -> out[((chunk*(N/16)+tile)*32+lane)*16 + e],
// zero-padded past kmax. W_gcn is packed as one [512,128] K-dim (layer
// boundaries coincide with chunk boundaries).
// =====================================================================
__global__ void k_pack_w16(const float* __restrict__ W,
                           _Float16* __restrict__ out, int kmax, int N,
                           int chunks) {
  const int idx = blockIdx.x * blockDim.x + threadIdx.x;
  const int ntiles = N / 16;
  if (idx >= chunks * ntiles * 32) return;
  const int lane = idx & 31;
  const int tile = (idx >> 5) % ntiles;
  const int ch   = (idx >> 5) / ntiles;
  const int n  = tile * 16 + (lane & 15);
  const int kh = ch * 32 + (lane >> 4) * 8;
  _Float16* o = out + ((size_t)idx << 4);
#pragma unroll
  for (int v = 0; v < 8; ++v) {
    const int k = kh + v;
    o[v]     = (k      < kmax) ? (_Float16)W[(size_t)k        * N + n] : (_Float16)0.0f;
    o[8 + v] = (k + 16 < kmax) ? (_Float16)W[(size_t)(k + 16) * N + n] : (_Float16)0.0f;
  }
}

// =====================================================================
// Kernel 1: node encoder  h_node16 = f16(x_g @ W_enc + b_enc); K=32 -> 1 WMMA
// =====================================================================
__global__ __launch_bounds__(256) void k_encode(
    const float* __restrict__ xg, const _Float16* __restrict__ Wp,
    const float* __restrict__ benc, _Float16* __restrict__ hnode16) {
  const int wave = threadIdx.x >> 5, lane = threadIdx.x & 31;
  const int m0 = blockIdx.x * 16, n0 = wave * 16;
  const int half = lane >> 4, m = lane & 15;
  const float* row = xg + (size_t)(m0 + m) * NODE_DIM;
  v16h a;
#pragma unroll
  for (int e = 0; e < 8; ++e) {
    a[e]     = (_Float16)row[half * 8 + e];
    a[8 + e] = (_Float16)row[16 + half * 8 + e];
  }
  const v16h b = frag_b_packed(Wp, wave, lane);
  v8f c = {};
  c = __builtin_amdgcn_wmma_f32_16x16x32_f16(false, a, false, b, (short)0, c,
                                             false, false);
  const int n = n0 + (lane & 15);
#pragma unroll
  for (int r = 0; r < 8; ++r) {
    const int mm = (lane >> 4) * 8 + r;
    hnode16[(size_t)(m0 + mm) * HDIM + n] = (_Float16)(c[r] + benc[n]);
  }
}

// =====================================================================
// Kernel 2: message MLP  h = [h_src | h_dst | edge_attr | x_lg] @ W_msg + b
// K=276 padded to 288 (9 WMMA chunks). The two gathered 256B h_node rows
// per edge are staged with CDNA5 async global->LDS b128 copies (ASYNCcnt);
// only the 64B tail uses regular LDS stores.
// =====================================================================
__global__ __launch_bounds__(256) void k_msg_gemm(
    const _Float16* __restrict__ hnode16, const float* __restrict__ eattr,
    const float* __restrict__ xlg, const int* __restrict__ src_g,
    const int* __restrict__ dst_g, const _Float16* __restrict__ Wp,
    const float* __restrict__ bmsg, float* __restrict__ h,
    _Float16* __restrict__ x16) {
  __shared__ __align__(32) _Float16 As[16 * 288];   // 16 rows x 576B
  const int e0 = blockIdx.x * 16;
  {
    // 256 threads = 16 rows x 16 parts of 16B; two async copies per thread
    const int r = threadIdx.x >> 4;
    const int p = (threadIdx.x & 15) * 16;   // byte offset within 256B row
    const int e = e0 + r;
    const unsigned lds_src = (unsigned)(size_t)(&As[0]) + r * 576 + p;
    const unsigned lds_dst = lds_src + 256;
    const unsigned long long ga_src =
        (unsigned long long)(size_t)(hnode16 + (size_t)src_g[e] * HDIM) + p;
    const unsigned long long ga_dst =
        (unsigned long long)(size_t)(hnode16 + (size_t)dst_g[e] * HDIM) + p;
    asm volatile(
        "global_load_async_to_lds_b128 %0, %2, off\n\t"
        "global_load_async_to_lds_b128 %1, %3, off"
        :
        : "v"(lds_src), "v"(lds_dst), "v"(ga_src), "v"(ga_dst)
        : "memory");
  }
  // tail: 32 f16 per row (16 edge_attr, 4 x_lg, 12 zero-pad)
  for (int t = threadIdx.x; t < 16 * 32; t += 256) {
    const int r = t >> 5, k = 256 + (t & 31);
    const int e = e0 + r;
    _Float16 val;
    if (k < 272)      val = (_Float16)eattr[(size_t)e * EDGE_DIM + (k - 256)];
    else if (k < 276) val = (_Float16)xlg[(size_t)e * NBASIS + (k - 272)];
    else              val = (_Float16)0.0f;
    As[r * 288 + k] = val;
  }
  asm volatile("s_wait_asynccnt 0x0" ::: "memory");
  __syncthreads();
  const int wave = threadIdx.x >> 5, lane = threadIdx.x & 31;
  v8f acc = {};
#pragma unroll
  for (int kc = 0; kc < 9; ++kc) {
    const v16h a = frag_a_lds(As, 288, kc * 32, lane);
    const v16h b = frag_b_packed(Wp, kc * 8 + wave, lane);
    acc = __builtin_amdgcn_wmma_f32_16x16x32_f16(false, a, false, b, (short)0,
                                                 acc, false, false);
  }
  const int n = wave * 16 + (lane & 15);
#pragma unroll
  for (int r = 0; r < 8; ++r) {
    const int mm = (lane >> 4) * 8 + r;
    const float val = acc[r] + bmsg[n];
    h[(size_t)(e0 + mm) * HDIM + n]   = val;
    x16[(size_t)(e0 + mm) * HDIM + n] = (_Float16)val;
  }
}

// =====================================================================
// Kernel 3: basis embeds (K=4 -> scalar FMA), f16 out (L2-resident)
// =====================================================================
__global__ void k_basis_embed(const float* __restrict__ basis,
                              const float* __restrict__ W,
                              const float* __restrict__ b,
                              _Float16* __restrict__ out, int rows) {
  const size_t idx = (size_t)blockIdx.x * blockDim.x + threadIdx.x;
  if (idx >= (size_t)rows * HDIM) return;
  const int r = (int)(idx / HDIM), c = (int)(idx % HDIM);
  float acc = b[c];
#pragma unroll
  for (int k = 0; k < 4; ++k) acc += basis[(size_t)r * 4 + k] * W[k * HDIM + c];
  out[idx] = (_Float16)acc;
}

// =====================================================================
// Kernel 4: fills / per-layer segment-buffer init (vectorized)
// =====================================================================
__global__ void k_fill_f32(float* __restrict__ p, float v, size_t n) {
  const size_t idx = (size_t)blockIdx.x * blockDim.x + threadIdx.x;
  if (idx < n) p[idx] = v;
}
__global__ void k_fill4(float4* __restrict__ p, float v, size_t n4) {
  const size_t idx = (size_t)blockIdx.x * blockDim.x + threadIdx.x;
  if (idx < n4) p[idx] = make_float4(v, v, v, v);
}
__global__ void k_init_seg(float4* __restrict__ mx, float4* __restrict__ den,
                           float4* __restrict__ num, size_t n4) {
  const size_t idx = (size_t)blockIdx.x * blockDim.x + threadIdx.x;
  if (idx >= n4) return;
  mx[idx]  = make_float4(NEG_BIG, NEG_BIG, NEG_BIG, NEG_BIG);
  den[idx] = make_float4(0.f, 0.f, 0.f, 0.f);
  num[idx] = make_float4(0.f, 0.f, 0.f, 0.f);
}

// =====================================================================
// Kernel 5: res+ block input  x16 = f16(relu(bn(h)))
// =====================================================================
__global__ void k_prep_x(const float* __restrict__ h,
                         const float* __restrict__ gamma_l,
                         const float* __restrict__ beta_l,
                         _Float16* __restrict__ x16) {
  const size_t idx = (size_t)blockIdx.x * blockDim.x + threadIdx.x;
  if (idx >= (size_t)E_G * HDIM) return;
  const int c = (int)(idx % HDIM);
  const float v = h[idx] * (BN_INV * gamma_l[c]) + beta_l[c];
  x16[idx] = (_Float16)fmaxf(v, 0.0f);
}

// =====================================================================
// Kernel 6/7: segment-softmax passes over E_LG edges. 16 threads x 8 f16
// cols per edge (16B vector loads, f16 tensors L2-resident), f32 atomics.
// m recomputed in pass 2 instead of materialized (saves 128MB traffic).
// =====================================================================
__global__ void k_seg_max(const _Float16* __restrict__ x16,
                          const _Float16* __restrict__ nb16,
                          const _Float16* __restrict__ eb16,
                          const int* __restrict__ src,
                          const int* __restrict__ dst,
                          const float* __restrict__ tval,
                          float* __restrict__ mx) {
  const size_t gid = (size_t)blockIdx.x * blockDim.x + threadIdx.x;
  const int i  = (int)(gid >> 4);
  const int c8 = (int)(gid & 15) * 8;
  if (i >= E_LG) return;
  if (i + 256 < E_LG)  // stream the angle-basis rows ahead of us
    __builtin_prefetch(eb16 + (size_t)(i + 256) * HDIM + c8, 0, 1);
  const int s = src[i], d = dst[i];
  const float t = tval[0];
  const v8h xv = *(const v8h*)(x16  + (size_t)s * HDIM + c8);
  const v8h nv = *(const v8h*)(nb16 + (size_t)s * HDIM + c8);
  const v8h ev = *(const v8h*)(eb16 + (size_t)i * HDIM + c8);
#pragma unroll
  for (int j = 0; j < 8; ++j) {
    const float m = fmaxf((float)xv[j] * (float)nv[j] * (float)ev[j], 0.0f) + EPS_MSG;
    atomicMaxF32(&mx[(size_t)d * HDIM + c8 + j], m * t);
  }
}

__global__ void k_seg_sum(const _Float16* __restrict__ x16,
                          const _Float16* __restrict__ nb16,
                          const _Float16* __restrict__ eb16,
                          const int* __restrict__ src,
                          const int* __restrict__ dst,
                          const float* __restrict__ tval,
                          const float* __restrict__ mx,
                          float* __restrict__ den, float* __restrict__ num) {
  const size_t gid = (size_t)blockIdx.x * blockDim.x + threadIdx.x;
  const int i  = (int)(gid >> 4);
  const int c8 = (int)(gid & 15) * 8;
  if (i >= E_LG) return;
  if (i + 256 < E_LG)
    __builtin_prefetch(eb16 + (size_t)(i + 256) * HDIM + c8, 0, 1);
  const int s = src[i], d = dst[i];
  const float t = tval[0];
  const v8h xv = *(const v8h*)(x16  + (size_t)s * HDIM + c8);
  const v8h nv = *(const v8h*)(nb16 + (size_t)s * HDIM + c8);
  const v8h ev = *(const v8h*)(eb16 + (size_t)i * HDIM + c8);
#pragma unroll
  for (int j = 0; j < 8; ++j) {
    const size_t o = (size_t)d * HDIM + c8 + j;
    const float m  = fmaxf((float)xv[j] * (float)nv[j] * (float)ev[j], 0.0f) + EPS_MSG;
    const float ex = __expf(m * t - mx[o]);
    atomicAdd(&den[o], ex);
    atomicAdd(&num[o], ex * m);
  }
}

// =====================================================================
// Kernel 8: update GEMM  h' = (x + num/den) @ W_gcn[l] + b (+ h residual)
// K=128 -> 4 WMMA chunks; A tile (x+agg) built in LDS as f16.
// =====================================================================
__global__ __launch_bounds__(256) void k_update(
    const _Float16* __restrict__ x16, const float* __restrict__ numb,
    const float* __restrict__ denb, const _Float16* __restrict__ Wp,
    const float* __restrict__ bg, float* __restrict__ h, int residual) {
  __shared__ __align__(32) _Float16 As[16 * HDIM];
  const int e0 = blockIdx.x * 16;
  for (int t = threadIdx.x; t < 16 * HDIM; t += 256) {
    const size_t ix = (size_t)e0 * HDIM + t;
    const float dn  = denb[ix];
    const float agg = numb[ix] / (dn > 0.0f ? dn : 1.0f);
    As[t] = (_Float16)((float)x16[ix] + agg);
  }
  __syncthreads();
  const int wave = threadIdx.x >> 5, lane = threadIdx.x & 31;
  v8f acc = {};
#pragma unroll
  for (int kc = 0; kc < 4; ++kc) {
    const v16h a = frag_a_lds(As, HDIM, kc * 32, lane);
    const v16h b = frag_b_packed(Wp, kc * 8 + wave, lane);
    acc = __builtin_amdgcn_wmma_f32_16x16x32_f16(false, a, false, b, (short)0,
                                                 acc, false, false);
  }
  const int n = wave * 16 + (lane & 15);
#pragma unroll
  for (int r = 0; r < 8; ++r) {
    const int mm = (lane >> 4) * 8 + r;
    const size_t o = (size_t)(e0 + mm) * HDIM + n;
    float val = acc[r] + bg[n];
    if (residual) val += h[o];
    h[o] = val;
  }
}

// =====================================================================
// Kernel 9: final BN + scatter back to graph nodes (atomic f32 add)
// =====================================================================
__global__ void k_node_scatter(const float* __restrict__ h,
                               const int* __restrict__ dst_g,
                               const float* __restrict__ gammaL,
                               const float* __restrict__ betaL,
                               float* __restrict__ node_emb) {
  const size_t gid = (size_t)blockIdx.x * blockDim.x + threadIdx.x;
  const int e  = (int)(gid >> 5);
  const int c4 = (int)(gid & 31) * 4;
  if (e >= E_G) return;
  const int d = dst_g[e];
#pragma unroll
  for (int j = 0; j < 4; ++j) {
    const int c = c4 + j;
    const float v = h[(size_t)e * HDIM + c] * (BN_INV * gammaL[c]) + betaL[c];
    atomicAdd(&node_emb[(size_t)d * HDIM + c], v);
  }
}

// =====================================================================
// Kernel 10: batched mean pool (sums + counts via atomics)
// =====================================================================
__global__ void k_pool(const float* __restrict__ node_emb,
                       const int* __restrict__ batch_idx,
                       float* __restrict__ gsum, float* __restrict__ counts) {
  const size_t gid = (size_t)blockIdx.x * blockDim.x + threadIdx.x;
  const int n  = (int)(gid >> 5);
  const int c4 = (int)(gid & 31) * 4;
  if (n >= N_G) return;
  const int b = batch_idx[n];
#pragma unroll
  for (int j = 0; j < 4; ++j)
    atomicAdd(&gsum[(size_t)b * HDIM + c4 + j],
              node_emb[(size_t)n * HDIM + c4 + j]);
  if (c4 == 0) atomicAdd(&counts[b], 1.0f);
}

// =====================================================================
// Kernel 11: prediction head (256x10, K=128 — scalar is fine)
// =====================================================================
__global__ void k_pred(const float* __restrict__ gsum,
                       const float* __restrict__ counts,
                       const float* __restrict__ Wp,
                       const float* __restrict__ bp, float* __restrict__ out) {
  const int idx = blockIdx.x * blockDim.x + threadIdx.x;
  if (idx >= NBATCH * NTASKS) return;
  const int b = idx / NTASKS, t = idx % NTASKS;
  const float cn = fmaxf(counts[b], 1.0f);
  float acc = bp[t];
#pragma unroll 8
  for (int k = 0; k < HDIM; ++k)
    acc += (gsum[(size_t)b * HDIM + k] / cn) * Wp[k * NTASKS + t];
  out[idx] = acc;
}

// =====================================================================
// Host launcher
// =====================================================================
extern "C" void kernel_launch(void* const* d_in, const int* in_sizes, int n_in,
                              void* d_out, int out_size, void* d_ws,
                              size_t ws_size, hipStream_t stream) {
  (void)in_sizes; (void)n_in; (void)out_size; (void)ws_size;
  // ---- inputs (setup_inputs order) ----
  const float* x_g        = (const float*)d_in[0];
  const float* eattr_g    = (const float*)d_in[1];
  const float* x_lg       = (const float*)d_in[2];
  const float* dist_basis = (const float*)d_in[3];
  const float* eattr_lg   = (const float*)d_in[4];
  const int*   ei_g       = (const int*)d_in[5];   // [2, E_G]
  const int*   ei_lg      = (const int*)d_in[6];   // [2, E_LG]
  const int*   batch_idx  = (const int*)d_in[7];
  const float* W_enc  = (const float*)d_in[8];
  const float* b_enc  = (const float*)d_in[9];
  const float* W_msg  = (const float*)d_in[10];
  const float* b_msg  = (const float*)d_in[11];
  const float* W_nb   = (const float*)d_in[12];
  const float* b_nb   = (const float*)d_in[13];
  const float* W_eb   = (const float*)d_in[14];
  const float* b_eb   = (const float*)d_in[15];
  const float* W_gcn  = (const float*)d_in[16];    // [L,H,H]
  const float* b_gcn  = (const float*)d_in[17];    // [L,H]
  const float* t_vals = (const float*)d_in[18];    // [L]
  const float* gamma  = (const float*)d_in[19];    // [L,H]
  const float* beta   = (const float*)d_in[20];    // [L,H]
  const float* W_pred = (const float*)d_in[21];
  const float* b_pred = (const float*)d_in[22];
  float* out = (float*)d_out;
  const int* src_g = ei_g;            const int* dst_g = ei_g + E_G;
  const int* src_l = ei_lg;           const int* dst_l = ei_lg + E_LG;

  // ---- workspace carve (256B aligned) ----
  char* ws = (char*)d_ws;
  size_t off = 0;
  auto carve = [&](size_t bytes) {
    char* p = ws + off;
    off += (bytes + 255) & ~(size_t)255;
    return p;
  };
  _Float16* hnode16  = (_Float16*)carve((size_t)N_G * HDIM * 2);
  float*    h        = (float*)   carve((size_t)E_G * HDIM * 4);
  _Float16* x16      = (_Float16*)carve((size_t)E_G * HDIM * 2);
  _Float16* nb16     = (_Float16*)carve((size_t)E_G * HDIM * 2);
  _Float16* eb16     = (_Float16*)carve((size_t)E_LG * HDIM * 2);
  float*    mx       = (float*)   carve((size_t)E_G * HDIM * 4);
  float*    den      = (float*)   carve((size_t)E_G * HDIM * 4);
  float*    num      = (float*)   carve((size_t)E_G * HDIM * 4);
  float*    node_emb = (float*)   carve((size_t)N_G * HDIM * 4);
  float*    gsum     = (float*)   carve((size_t)NBATCH * HDIM * 4);
  float*    counts   = (float*)   carve((size_t)NBATCH * 4);
  // packed-fragment weights: frag = 32 lanes x 16 f16
  _Float16* Wp_enc = (_Float16*)carve((size_t)1  * 8 * 32 * 16 * 2);
  _Float16* Wp_msg = (_Float16*)carve((size_t)9  * 8 * 32 * 16 * 2);
  _Float16* Wp_gcn = (_Float16*)carve((size_t)16 * 8 * 32 * 16 * 2);

  const size_t nEH   = (size_t)E_G * HDIM;
  const int segGrid  = (int)(((size_t)E_LG * (HDIM / 8) + 255) / 256);
  const int ehGrid   = (int)((nEH + 255) / 256);

  // ---- one-time weight packing (tiny) ----
  k_pack_w16<<<(1 * 8 * 32 + 255) / 256, 256, 0, stream>>>(W_enc, Wp_enc,
                                                           NODE_DIM, HDIM, 1);
  k_pack_w16<<<(9 * 8 * 32 + 255) / 256, 256, 0, stream>>>(W_msg, Wp_msg, 276,
                                                           HDIM, 9);
  k_pack_w16<<<(16 * 8 * 32 + 255) / 256, 256, 0, stream>>>(
      W_gcn, Wp_gcn, NLAYER * HDIM, HDIM, 16);   // [512,128] view

  // ---- encoder + message MLP + basis embeds ----
  k_encode<<<N_G / 16, 256, 0, stream>>>(x_g, Wp_enc, b_enc, hnode16);
  k_msg_gemm<<<E_G / 16, 256, 0, stream>>>(hnode16, eattr_g, x_lg, src_g,
                                           dst_g, Wp_msg, b_msg, h, x16);
  k_basis_embed<<<ehGrid, 256, 0, stream>>>(dist_basis, W_nb, b_nb, nb16, E_G);
  k_basis_embed<<<(int)(((size_t)E_LG * HDIM + 255) / 256), 256, 0, stream>>>(
      eattr_lg, W_eb, b_eb, eb16, E_LG);

  // ---- 4 GENConv res+ layers ----
  for (int l = 0; l < NLAYER; ++l) {
    if (l > 0)
      k_prep_x<<<ehGrid, 256, 0, stream>>>(h, gamma + (size_t)(l - 1) * HDIM,
                                           beta + (size_t)(l - 1) * HDIM, x16);
    k_init_seg<<<(int)((nEH / 4 + 255) / 256), 256, 0, stream>>>(
        (float4*)mx, (float4*)den, (float4*)num, nEH / 4);
    k_seg_max<<<segGrid, 256, 0, stream>>>(x16, nb16, eb16, src_l, dst_l,
                                           t_vals + l, mx);
    k_seg_sum<<<segGrid, 256, 0, stream>>>(x16, nb16, eb16, src_l, dst_l,
                                           t_vals + l, mx, den, num);
    k_update<<<E_G / 16, 256, 0, stream>>>(
        x16, num, den, Wp_gcn + (size_t)l * 4 * 8 * 32 * 16,
        b_gcn + (size_t)l * HDIM, h, l > 0 ? 1 : 0);
  }

  // ---- final BN + node scatter + mean pool + head ----
  k_fill4<<<(int)(((size_t)N_G * HDIM / 4 + 255) / 256), 256, 0, stream>>>(
      (float4*)node_emb, 0.0f, (size_t)N_G * HDIM / 4);
  k_fill4<<<(NBATCH * HDIM / 4 + 255) / 256, 256, 0, stream>>>(
      (float4*)gsum, 0.0f, (size_t)NBATCH * HDIM / 4);
  k_fill_f32<<<1, 256, 0, stream>>>(counts, 0.0f, (size_t)NBATCH);
  k_node_scatter<<<(int)(((size_t)E_G * (HDIM / 4) + 255) / 256), 256, 0,
                   stream>>>(h, dst_g, gamma + (size_t)(NLAYER - 1) * HDIM,
                             beta + (size_t)(NLAYER - 1) * HDIM, node_emb);
  k_pool<<<(int)(((size_t)N_G * (HDIM / 4) + 255) / 256), 256, 0, stream>>>(
      node_emb, batch_idx, gsum, counts);
  k_pred<<<(NBATCH * NTASKS + 255) / 256, 256, 0, stream>>>(gsum, counts,
                                                            W_pred, b_pred, out);
}
